// MoE_55740085567780
// MI455X (gfx1250) — matrix operand order repaired and checked by reading, not compiled, not run on hardware
//
#include <hip/hip_runtime.h>

// MoE: B=8, S=4096, H=512, E=8, top-2.
// Routed (top-2) grouped GEMM with bf16 WMMA (v_wmma_f32_16x16x32_bf16),
// f32 accumulate, f32 gating/softmax, f32 atomic combine.
// M-tile = 32 tokens/block (2 WMMA M-subtiles) so each B-operand register load
// feeds two WMMAs -> halves L2 weight traffic vs M=16.

#define T_TOKENS 32768   // B*S
#define HD 512           // H
#define FD 1024          // 2H
#define NE 8             // experts
#define MT 32            // tokens per block (2 x 16-row WMMA tiles)

typedef __attribute__((ext_vector_type(16))) __bf16 v16bf;
typedef __attribute__((ext_vector_type(8)))  float  v8f;

union V16U { uint4 u[2]; v16bf v; };

// ---------------------------------------------------------------- zero init
__global__ void moe_zero_kernel(float4* __restrict__ out4, long n4,
                                int* __restrict__ counts) {
    long i = (long)blockIdx.x * blockDim.x + threadIdx.x;
    if (i < n4) out4[i] = make_float4(0.f, 0.f, 0.f, 0.f);
    if (i < NE) counts[i] = 0;
}

// ------------------------------------------- weight transpose + bf16 convert
// src layout [E][P][Q] (f32)  ->  dst layout [E][Q][P] (bf16)
__global__ void moe_transpose_bf16_kernel(const float* __restrict__ src,
                                          __bf16* __restrict__ dst,
                                          int P, int Q, long total) {
    long i = (long)blockIdx.x * blockDim.x + threadIdx.x;
    if (i >= total) return;
    int  p = (int)(i % P);
    long r = i / P;
    int  q = (int)(r % Q);
    long e = r / Q;
    dst[i] = (__bf16)src[(e * P + p) * (long)Q + q];
}

// ---------------------------------------------------------------- gating
__global__ void moe_gate_kernel(const float* __restrict__ x,
                                const float* __restrict__ gw,
                                const float* __restrict__ gb,
                                int* __restrict__ counts,
                                int* __restrict__ tok_list,
                                float* __restrict__ w_list) {
    int t = blockIdx.x * blockDim.x + threadIdx.x;
    if (t >= T_TOKENS) return;

    float acc[NE];
#pragma unroll
    for (int e = 0; e < NE; ++e) acc[e] = gb[e];

    const float* xr = x + (size_t)t * HD;
    for (int h = 0; h < HD; ++h) {
        float xv = xr[h];
#pragma unroll
        for (int e = 0; e < NE; ++e) acc[e] += xv * gw[h * NE + e];
    }

    // top-1 (ties -> lowest index, matching lax.top_k)
    int i0 = 0; float v0 = acc[0];
#pragma unroll
    for (int e = 1; e < NE; ++e) if (acc[e] > v0) { v0 = acc[e]; i0 = e; }
    // top-2
    int i1 = -1; float v1 = -INFINITY;
#pragma unroll
    for (int e = 0; e < NE; ++e)
        if (e != i0 && acc[e] > v1) { v1 = acc[e]; i1 = e; }

    // softmax over the two selected logits (v0 >= v1)
    float e1  = expf(v1 - v0);
    float inv = 1.0f / (1.0f + e1);
    float w0  = inv;
    float w1  = e1 * inv;

    int p0 = atomicAdd(&counts[i0], 1);
    tok_list[i0 * T_TOKENS + p0] = t;
    w_list [i0 * T_TOKENS + p0] = w0;
    int p1 = atomicAdd(&counts[i1], 1);
    tok_list[i1 * T_TOKENS + p1] = t;
    w_list [i1 * T_TOKENS + p1] = w1;
}

// ---------------------------------------------------------------- WMMA helpers
// A operand (16x32 bf16): lane row m = lane&15; lanes 0-15 carry K {0..15},
// lanes 16-31 carry K {16..31} (matches B). Two contiguous 16B LDS loads.
__device__ __forceinline__ v16bf moe_load_A(const __bf16* row, int k0, int kbA) {
    V16U a;
    a.u[0] = *reinterpret_cast<const uint4*>(row + k0 + kbA);
    a.u[1] = *reinterpret_cast<const uint4*>(row + k0 + 8 + kbA);
    return a.v;
}
// B operand (32x16 bf16): lane col n = lane&15; element i -> K = 16*(lane>>4)+i.
// With W stored N-major ([n][k]), this is 32 contiguous bytes.
__device__ __forceinline__ v16bf moe_load_B(const __bf16* colk) {
    V16U b;
    const uint4* p = reinterpret_cast<const uint4*>(colk);
    b.u[0] = p[0];
    b.u[1] = p[1];
    return b.v;
}

// ---------------------------------------------------------------- expert GEMM
// grid: (T_TOKENS/MT, NE), block: 512 (16 waves). One block = 32 routed tokens
// of one expert: X[32,512] -> relu(X@W1+b1)[32,1024] -> (@W2+b2)[32,512]
// -> *route_weight -> atomic add into out.
__global__ __launch_bounds__(512) void moe_expert_kernel(
        const float*  __restrict__ x,
        const __bf16* __restrict__ w1t,   // [E][FD][HD]  (N-major)
        const float*  __restrict__ b1,    // [E][FD]
        const __bf16* __restrict__ w2t,   // [E][HD][FD]  (N-major)
        const float*  __restrict__ b2,    // [E][HD]
        const int*    __restrict__ counts,
        const int*    __restrict__ tok_list,
        const float*  __restrict__ w_list,
        float*        __restrict__ out) {

    __shared__ __bf16 Xs[MT * 520];    // 32 rows x 512 (+8 pad)  = 33,280 B
    __shared__ __bf16 Hs[MT * 1032];   // 32 rows x 1024 (+8 pad) = 66,048 B
    __shared__ int    s_tok[MT];
    __shared__ float  s_w[MT];

    const int e    = blockIdx.y;
    const int base = blockIdx.x * MT;
    const int ntok = counts[e];
    if (base >= ntok) return;          // uniform exit: no barrier yet

    const int tid  = threadIdx.x;
    const int lane = tid & 31;
    const int wv   = tid >> 5;         // 0..15
    const int nlo  = lane & 15;
    const int hi   = lane >> 4;
    const int kbA  = hi * 16;          // A half-lane K offset
    const int kbB  = hi * 16;          // B half-lane K offset

    if (tid < MT) {
        int idx = base + tid;
        int tok = -1; float wgt = 0.f;
        if (idx < ntok) {
            tok = tok_list[e * T_TOKENS + idx];
            wgt = w_list [e * T_TOKENS + idx];
        }
        s_tok[tid] = tok;
        s_w[tid]   = wgt;
    }
    __syncthreads();

    // Gather 32 token rows, convert f32 -> bf16 into LDS.
    for (int c = tid; c < MT * 128; c += 512) {
        int r  = c >> 7;
        int cc = c & 127;
        int tok = s_tok[r];
        float4 v = make_float4(0.f, 0.f, 0.f, 0.f);
        if (tok >= 0)
            v = reinterpret_cast<const float4*>(x + (size_t)tok * HD)[cc];
        __bf16* d = &Xs[r * 520 + cc * 4];
        d[0] = (__bf16)v.x; d[1] = (__bf16)v.y;
        d[2] = (__bf16)v.z; d[3] = (__bf16)v.w;
    }
    __syncthreads();

    // ---------------- FC1: H1[32,1024] = relu(X @ W1 + b1) ------------------
    // wave wv owns 4 N-tiles [wv*64, wv*64+64); each B tile feeds 2 M-subtiles.
    const v8f vzero = {};
    v8f acc1[4][2];
#pragma unroll
    for (int t = 0; t < 4; ++t) { acc1[t][0] = vzero; acc1[t][1] = vzero; }

    const int n0 = wv * 64;
    const __bf16* w1base = w1t + (size_t)e * FD * HD;
    const __bf16* xrow0  = &Xs[nlo * 520];          // A rows 0..15
    const __bf16* xrow1  = &Xs[(16 + nlo) * 520];   // A rows 16..31

    for (int kk = 0; kk < 16; ++kk) {                // K = 512 -> 16 steps
        const int k0 = kk * 32;
        v16bf a0 = moe_load_A(xrow0, k0, kbA);
        v16bf a1 = moe_load_A(xrow1, k0, kbA);
        v16bf b[4];
#pragma unroll
        for (int t = 0; t < 4; ++t) {               // load phase (one clause)
            const int n = n0 + t * 16 + nlo;
            b[t] = moe_load_B(w1base + (size_t)n * HD + k0 + kbB);
        }
#pragma unroll
        for (int t = 0; t < 4; ++t) {               // compute phase
            acc1[t][0] = __builtin_amdgcn_wmma_f32_16x16x32_bf16(
                false, a0, false, b[t], (short)0, acc1[t][0], false, false);
            acc1[t][1] = __builtin_amdgcn_wmma_f32_16x16x32_bf16(
                false, a1, false, b[t], (short)0, acc1[t][1], false, false);
        }
    }
#pragma unroll
    for (int t = 0; t < 4; ++t) {
        const int n  = n0 + t * 16 + nlo;
        const float bias = b1[e * FD + n];
#pragma unroll
        for (int s = 0; s < 2; ++s) {
#pragma unroll
            for (int v = 0; v < 8; ++v) {            // C: M = s*16 + v + 8*hi
                const int m = s * 16 + v + 8 * hi;
                float val = acc1[t][s][v] + bias;
                val = val > 0.f ? val : 0.f;
                Hs[m * 1032 + n] = (__bf16)val;
            }
        }
    }
    __syncthreads();

    // ---------------- FC2: Y[32,512] = H1 @ W2 + b2 -------------------------
    // wave wv owns 2 N-tiles [wv*32, wv*32+32).
    v8f acc2[2][2];
#pragma unroll
    for (int j = 0; j < 2; ++j) { acc2[j][0] = vzero; acc2[j][1] = vzero; }

    const int n0b = wv * 32;
    const __bf16* w2base = w2t + (size_t)e * HD * FD;
    const __bf16* hrow0  = &Hs[nlo * 1032];
    const __bf16* hrow1  = &Hs[(16 + nlo) * 1032];

    for (int kk = 0; kk < 32; ++kk) {                // K = 1024 -> 32 steps
        const int k0 = kk * 32;
        v16bf a0 = moe_load_A(hrow0, k0, kbA);
        v16bf a1 = moe_load_A(hrow1, k0, kbA);
        v16bf b[2];
#pragma unroll
        for (int j = 0; j < 2; ++j) {
            const int n = n0b + j * 16 + nlo;
            b[j] = moe_load_B(w2base + (size_t)n * FD + k0 + kbB);
        }
#pragma unroll
        for (int j = 0; j < 2; ++j) {
            acc2[j][0] = __builtin_amdgcn_wmma_f32_16x16x32_bf16(
                false, a0, false, b[j], (short)0, acc2[j][0], false, false);
            acc2[j][1] = __builtin_amdgcn_wmma_f32_16x16x32_bf16(
                false, a1, false, b[j], (short)0, acc2[j][1], false, false);
        }
    }
#pragma unroll
    for (int j = 0; j < 2; ++j) {
        const int n  = n0b + j * 16 + nlo;
        const float bias = b2[e * HD + n];
#pragma unroll
        for (int s = 0; s < 2; ++s) {
#pragma unroll
            for (int v = 0; v < 8; ++v) {
                const int m   = s * 16 + v + 8 * hi;
                const int tok = s_tok[m];
                if (tok >= 0) {
                    float val = (acc2[j][s][v] + bias) * s_w[m];
                    atomicAdd(out + (size_t)tok * HD + n, val);
                }
            }
        }
    }
}

// ---------------------------------------------------------------- launcher
extern "C" void kernel_launch(void* const* d_in, const int* in_sizes, int n_in,
                              void* d_out, int out_size, void* d_ws, size_t ws_size,
                              hipStream_t stream) {
    const float* x      = (const float*)d_in[0];  // [8,4096,512]
    const float* gate_w = (const float*)d_in[1];  // [512,8]
    const float* gate_b = (const float*)d_in[2];  // [8]
    const float* w1     = (const float*)d_in[3];  // [8,512,1024]
    const float* b1     = (const float*)d_in[4];  // [8,1024]
    const float* w2     = (const float*)d_in[5];  // [8,1024,512]
    const float* b2     = (const float*)d_in[6];  // [8,512]
    float* out = (float*)d_out;                   // [8,4096,512]

    // workspace layout (~19 MB)
    char* ws = (char*)d_ws;
    __bf16* w1t      = (__bf16*)(ws);                         //  8,388,608 B
    __bf16* w2t      = (__bf16*)(ws + 8388608);               //  8,388,608 B
    int*    counts   = (int*)   (ws + 16777216);              //        256 B (pad)
    int*    tok_list = (int*)   (ws + 16777216 + 256);        //  1,048,576 B
    float*  w_list   = (float*) (ws + 16777216 + 256 + 4 * NE * T_TOKENS);

    // 1) zero output + counters
    {
        long n4 = (long)T_TOKENS * HD / 4;   // 4,194,304 float4
        moe_zero_kernel<<<(unsigned)((n4 + 255) / 256), 256, 0, stream>>>(
            (float4*)out, n4, counts);
    }
    // 2) weight transpose + bf16 convert
    {
        long total = (long)NE * HD * FD;     // 4,194,304
        moe_transpose_bf16_kernel<<<(unsigned)((total + 255) / 256), 256, 0, stream>>>(
            w1, w1t, /*P=*/HD, /*Q=*/FD, total);   // [E][H][F] -> [E][F][H]
        moe_transpose_bf16_kernel<<<(unsigned)((total + 255) / 256), 256, 0, stream>>>(
            w2, w2t, /*P=*/FD, /*Q=*/HD, total);   // [E][F][H] -> [E][H][F]
    }
    // 3) gating: top-2 + softmax + expert lists
    moe_gate_kernel<<<T_TOKENS / 256, 256, 0, stream>>>(
        x, gate_w, gate_b, counts, tok_list, w_list);
    // 4) routed expert GEMMs (blocks past each expert's count exit immediately)
    {
        dim3 grid(T_TOKENS / MT, NE);
        moe_expert_kernel<<<grid, 512, 0, stream>>>(
            x, w1t, b1, w2t, b2, counts, tok_list, w_list, out);
    }
}
